// DiscreteMMSE_55216099558210
// MI455X (gfx1250) — matrix operand order repaired and checked by reading, not compiled, not run on hardware
//
#include <hip/hip_runtime.h>
#include <cmath>

#define B_SZ   128
#define N_PTS  128
#define D_DIM  64
#define T_DIM  4096
#define TC     128                 // T-chunk per iteration
#define NCH    (T_DIM / TC)        // 32 chunks
#define HALF_LOG_2PI 0.9189385332046727f

typedef __attribute__((ext_vector_type(16))) _Float16 v16h;
typedef __attribute__((ext_vector_type(8)))  _Float16 v8h;
typedef __attribute__((ext_vector_type(8)))  float    v8f;
typedef __attribute__((ext_vector_type(4)))  int      v4i;

// ---------------- CDNA5 async global->LDS (guarded) ----------------
#if defined(__gfx1250__) && __has_builtin(__builtin_amdgcn_global_load_async_to_lds_b128) && __has_builtin(__builtin_amdgcn_s_wait_asynccnt)
#define HAVE_ASYNC_LDS 1
#else
#define HAVE_ASYNC_LDS 0
#endif

__device__ __forceinline__ void cp16(_Float16* lds_dst, const _Float16* gsrc) {
#if HAVE_ASYNC_LDS
  __builtin_amdgcn_global_load_async_to_lds_b128(
      (v4i __attribute__((address_space(1)))*)(gsrc),
      (v4i __attribute__((address_space(3)))*)(lds_dst), 0, 0);
#else
  *(v8h*)lds_dst = *(const v8h*)gsrc;   // 16-byte flat store to LDS
#endif
}

__device__ __forceinline__ void async_copy_wait() {
#if HAVE_ASYNC_LDS
  __builtin_amdgcn_s_wait_asynccnt(0);
#endif
}

// ---------------- WMMA helpers ----------------
__device__ __forceinline__ v8f wmma16(v16h a, v16h b, v8f c) {
  // D = A(16x32 f16) * B(32x16 f16) + C(f32)
  return __builtin_amdgcn_wmma_f32_16x16x32_f16(
      false, a, false, b, (short)0, c, false, false);
}

// A-fragment: halves [0..7] from p, halves [8..15] from p+16 (K interleave per ISA)
__device__ __forceinline__ v16h ldA(const _Float16* p) {
  v8h lo = *(const v8h*)p;
  v8h hi = *(const v8h*)(p + 16);
  v16h r;
#pragma unroll
  for (int i = 0; i < 8; ++i) { r[i] = lo[i]; r[8 + i] = hi[i]; }
  return r;
}

// B-fragment: halves [0..7] from p, halves [8..15] from p+16 (K interleave per ISA)
__device__ __forceinline__ v16h ldB(const _Float16* p) {
  v8h lo = *(const v8h*)p;
  v8h hi = *(const v8h*)(p + 16);
  v16h r;
#pragma unroll
  for (int i = 0; i < 8; ++i) { r[i] = lo[i]; r[8 + i] = hi[i]; }
  return r;
}

// ---------------- pre-pass: f16 conversions + transposed W + W.mean ----------------
__global__ void dmmse_convert(const float* __restrict__ data, const float* __restrict__ W,
                              _Float16* __restrict__ Xh, _Float16* __restrict__ Wh,
                              _Float16* __restrict__ Wt) {
  int i = blockIdx.x * blockDim.x + threadIdx.x;
  const int nX = B_SZ * N_PTS * D_DIM;
  if (i < nX) Xh[i] = (_Float16)data[i];
  const int nW = D_DIM * T_DIM;
  if (i < nW) {
    float w = W[i];
    Wh[i] = (_Float16)w;                       // (d, t): K=t contiguous  -> GEMM2 B
    int d = i >> 12, t = i & (T_DIM - 1);
    Wt[t * D_DIM + d] = (_Float16)w;           // (t, d): K=d contiguous  -> GEMM1 B
  }
}

__global__ void dmmse_wmean(const float* __restrict__ W, float* __restrict__ wmean) {
  int d = threadIdx.x;                          // 64 threads
  float s = 0.f;
  for (int t = 0; t < T_DIM; ++t) s += W[(size_t)d * T_DIM + t];
  wmean[d] = s * (1.0f / (float)T_DIM);
}

// ---------------- fused flash-style kernel: one block per batch b ----------------
extern "C" __global__ void __launch_bounds__(256)
dmmse_main(const float* __restrict__ data, const float* __restrict__ targets,
           const _Float16* __restrict__ Xh, const _Float16* __restrict__ Wh,
           const _Float16* __restrict__ Wt, const float* __restrict__ wmean,
           float* __restrict__ out) {
  extern __shared__ char smem[];
  _Float16* WtL = (_Float16*)smem;                  // TC x 64   (t-major)  16 KB
  _Float16* WhL = WtL + TC * D_DIM;                 // 64 x TC   (d-major)  16 KB
  float*    Ef  = (float*)(WhL + D_DIM * TC);       // N x TC f32           64 KB
  _Float16* Ph  = (_Float16*)(Ef + N_PTS * TC);     // N x TC f16           32 KB
  float*    sL  = (float*)(Ph + N_PTS * TC);        // N f32 rescale        0.5 KB

  const int tid   = threadIdx.x;
  const int lane  = tid & 31;
  const int wv    = tid >> 5;           // 8 waves; wave owns 16 rows
  const int rbase = wv * 16;
  const int hio   = (lane & 16) ? 8 : 0;   // A-frag K offset / D-row offset, hi half-wave
  const int klo   = (lane & 16) ? 8 : 0;   // B-frag K offset, hi half-wave
  const int b     = blockIdx.x;

  // GEMM1 A-fragments (this wave's 16 data rows, K=d=0..63) — loaded once
  const _Float16* xrow = Xh + ((size_t)b * N_PTS + rbase + (lane & 15)) * D_DIM;
  v16h a0 = ldA(xrow + hio);           // K 0..31
  v16h a1 = ldA(xrow + 32 + hio);      // K 32..63

  float yv[8];
#pragma unroll
  for (int r = 0; r < 8; ++r)
    yv[r] = targets[(size_t)b * N_PTS + rbase + r + hio];

  v8f acc[4] = {};                      // 16 rows x 64 cols of w_mmse accumulator
  float m_run = -INFINITY, l_run = 0.0f;

  for (int ch = 0; ch < NCH; ++ch) {
    const int t0 = ch * TC;

    // ---- stage W chunks into LDS (async if available) ----
    {
      const _Float16* src = Wt + (size_t)t0 * D_DIM;     // contiguous 16 KB
      for (int i = tid; i < TC * D_DIM / 8; i += 256)
        cp16(WtL + i * 8, src + i * 8);
      for (int i = tid; i < D_DIM * TC / 8; i += 256) {  // 64 strided rows
        int d = i / (TC / 8), t8 = i % (TC / 8);
        cp16(WhL + d * TC + t8 * 8, Wh + (size_t)d * T_DIM + t0 + t8 * 8);
      }
      async_copy_wait();
      if (ch + 1 < NCH)                                   // global_prefetch_b8
        __builtin_prefetch(Wt + (size_t)(t0 + TC) * D_DIM + tid * 64, 0, 0);
    }
    __syncthreads();

    // ---- GEMM1: E = X * Wchunk ; write logp into Ef ----
#pragma unroll
    for (int ct = 0; ct < TC / 16; ++ct) {
      const _Float16* bp = WtL + (ct * 16 + (lane & 15)) * D_DIM + klo;
      v16h b0 = ldB(bp);        // K = d 0..31
      v16h b1 = ldB(bp + 32);   // K = d 32..63
      v8f c = {};
      c = wmma16(a0, b0, c);
      c = wmma16(a1, b1, c);
#pragma unroll
      for (int r = 0; r < 8; ++r) {
        float e = yv[r] - c[r];
        Ef[(rbase + r + hio) * TC + ct * 16 + (lane & 15)] =
            -0.5f * e * e - HALF_LOG_2PI;
      }
    }
    __syncthreads();

    // ---- prefix-sum over n (one thread per t-column) ----
    if (tid < TC) {
      float run = 0.0f;
      for (int n = 0; n < N_PTS; ++n) {
        run += Ef[n * TC + tid];
        Ef[n * TC + tid] = run;
      }
    }
    __syncthreads();

    // ---- online softmax per row over this chunk's T columns ----
    if (tid < N_PTS) {
      const int row = tid;
      float cm = -INFINITY;
      for (int c = 0; c < TC; ++c) {
        int cc = (c + lane) & (TC - 1);        // bank-rotated
        cm = fmaxf(cm, Ef[row * TC + cc]);
      }
      float mn = fmaxf(m_run, cm);
      float s  = __expf(m_run - mn);
      float ps = 0.0f;
      for (int c = 0; c < TC; ++c) {
        int cc = (c + lane) & (TC - 1);
        float p = __expf(Ef[row * TC + cc] - mn);
        Ph[row * TC + cc] = (_Float16)p;
        ps += p;
      }
      l_run = l_run * s + ps;
      m_run = mn;
      sL[row] = s;
    }
    __syncthreads();

    // ---- GEMM2: acc = acc * s(row) + P * Wchunk^T ----
    float s8[8];
#pragma unroll
    for (int r = 0; r < 8; ++r) s8[r] = sL[rbase + r + hio];
#pragma unroll
    for (int dt = 0; dt < 4; ++dt)
#pragma unroll
      for (int r = 0; r < 8; ++r) acc[dt][r] *= s8[r];

#pragma unroll
    for (int kk = 0; kk < TC / 32; ++kk) {
      const _Float16* pa = Ph + (rbase + (lane & 15)) * TC + kk * 32 + hio;
      v16h pA = ldA(pa);
#pragma unroll
      for (int dt = 0; dt < 4; ++dt) {
        const _Float16* pb = WhL + (dt * 16 + (lane & 15)) * TC + kk * 32 + klo;
        acc[dt] = wmma16(pA, ldB(pb), acc[dt]);
      }
    }
    __syncthreads();   // protect LDS before next chunk's staging
  }

  // ---- finalize: w_mmse = acc / l ; preds ----
  float* Wf = Ef;                        // reuse: N x 64 f32
#pragma unroll
  for (int dt = 0; dt < 4; ++dt)
#pragma unroll
    for (int r = 0; r < 8; ++r)
      Wf[(rbase + r + hio) * D_DIM + dt * 16 + (lane & 15)] = acc[dt][r];
  __syncthreads();

  if (tid < N_PTS - 1) {
    const float inv = 1.0f / l_run;      // this thread owned row 'tid' in softmax
    const float* xr = data + ((size_t)b * N_PTS + tid + 1) * D_DIM;
    float p = 0.0f;
    for (int d = 0; d < D_DIM; ++d) p += xr[d] * Wf[tid * D_DIM + d];
    out[(size_t)b * N_PTS + tid + 1] = p * inv;
  } else if (tid == N_PTS - 1) {
    const float* x0 = data + (size_t)b * N_PTS * D_DIM;
    float p = 0.0f;
    for (int d = 0; d < D_DIM; ++d) p += x0[d] * wmean[d];
    out[(size_t)b * N_PTS] = p;          // prior-mean prediction
  }
}

// ---------------- launcher ----------------
extern "C" void kernel_launch(void* const* d_in, const int* in_sizes, int n_in,
                              void* d_out, int out_size, void* d_ws, size_t ws_size,
                              hipStream_t stream) {
  (void)in_sizes; (void)n_in; (void)out_size; (void)ws_size;
  const float* data    = (const float*)d_in[0];
  const float* targets = (const float*)d_in[1];
  const float* W       = (const float*)d_in[2];
  float* out = (float*)d_out;

  char* ws = (char*)d_ws;
  _Float16* Xh = (_Float16*)ws;                                   // 2 MB
  _Float16* Wh = Xh + (size_t)B_SZ * N_PTS * D_DIM;               // 512 KB
  _Float16* Wt = Wh + (size_t)D_DIM * T_DIM;                      // 512 KB
  float*    wmean = (float*)(Wt + (size_t)D_DIM * T_DIM);         // 256 B

  const int nconv = B_SZ * N_PTS * D_DIM;
  dmmse_convert<<<(nconv + 255) / 256, 256, 0, stream>>>(data, W, Xh, Wh, Wt);
  dmmse_wmean<<<1, 64, 0, stream>>>(W, wmean);

  const size_t shmem = (size_t)TC * D_DIM * 2      // WtL
                     + (size_t)D_DIM * TC * 2      // WhL
                     + (size_t)N_PTS * TC * 4      // Ef
                     + (size_t)N_PTS * TC * 2      // Ph
                     + (size_t)N_PTS * 4;          // sL   ≈ 131.5 KB
  dmmse_main<<<dim3(B_SZ), dim3(256), shmem, stream>>>(
      data, targets, Xh, Wh, Wt, wmean, out);
}
